// MultiHeadAttention_89824946029143
// MI455X (gfx1250) — matrix operand order repaired and checked by reading, not compiled, not run on hardware
//
#include <hip/hip_runtime.h>

typedef __attribute__((ext_vector_type(16))) __bf16 v16bf;
typedef __attribute__((ext_vector_type(8)))  __bf16 v8bf;
typedef __attribute__((ext_vector_type(8)))  float  v8f;
typedef __attribute__((ext_vector_type(4)))  float  v4f;

#define D_MODEL   1024
#define NUM_HEADS 16
#define HEAD_DIM  64
#define BATCH     2
#define SEQ       2048
#define MROWS     (BATCH * SEQ)   // 4096

// clang lowers fptrunc f32->bf16 to v_cvt_pk_bf16_f32 on gfx1250
static __device__ __forceinline__ __bf16 f2bf(float f) { return (__bf16)f; }

// ---------------------------------------------------------------------------
// Streaming f32 -> bf16 conversion (8 elements / thread).
// ---------------------------------------------------------------------------
__global__ __launch_bounds__(256)
void mha_cvt_kernel(const float* __restrict__ s, __bf16* __restrict__ d) {
  const size_t i = ((size_t)blockIdx.x * blockDim.x + threadIdx.x) * 8;
  v4f a = *(const v4f*)(s + i);
  v4f b = *(const v4f*)(s + i + 4);
  v8bf pk;
  #pragma unroll
  for (int j = 0; j < 4; ++j) { pk[j] = f2bf(a[j]); pk[4 + j] = f2bf(b[j]); }
  *(v8bf*)(d + i) = pk;
}

// ---------------------------------------------------------------------------
// QKV projection: out = x @ W^T + b   (x bf16 [4096,1024], W bf16 [1024,1024])
// One wave computes a 32x64 strip (2 m-subtiles x 4 n-tiles): each B fragment
// feeds two WMMAs, halving weight traffic per FLOP.
// vmode 0: out [B,H,S,HD] (Q,K);  vmode 1: out [B,H,HD,S] (V^T)
// ---------------------------------------------------------------------------
__global__ __launch_bounds__(256)
void mha_qkv_proj_kernel(const __bf16* __restrict__ x, const __bf16* __restrict__ w,
                         const float* __restrict__ bias, __bf16* __restrict__ out,
                         int vmode) {
  const int lane = threadIdx.x & 31;
  const int g    = lane >> 4;
  const int ln   = lane & 15;
  const int wid  = (blockIdx.x * blockDim.x + threadIdx.x) >> 5;  // 0..2047
  const int m0   = (wid >> 4) * 32;   // 128 m-groups of 32 rows
  const int n0   = (wid & 15) * 64;   // 16 n-strips

  v8f acc[2][4];
  #pragma unroll
  for (int t = 0; t < 4; ++t) {
    const float bv = bias[n0 + t * 16 + ln];
    #pragma unroll
    for (int r = 0; r < 8; ++r) { acc[0][t][r] = bv; acc[1][t][r] = bv; }
  }

  const __bf16* xrow0 = x + (size_t)(m0 + ln) * D_MODEL;
  const __bf16* xrow1 = x + (size_t)(m0 + 16 + ln) * D_MODEL;

  for (int k0 = 0; k0 < D_MODEL; k0 += 32) {
    v8bf l0 = *(const v8bf*)(xrow0 + k0 + 16 * g);
    v8bf h0 = *(const v8bf*)(xrow0 + k0 + 16 * g + 8);
    v8bf l1 = *(const v8bf*)(xrow1 + k0 + 16 * g);
    v8bf h1 = *(const v8bf*)(xrow1 + k0 + 16 * g + 8);
    v16bf a0 = __builtin_shufflevector(l0, h0, 0,1,2,3,4,5,6,7,8,9,10,11,12,13,14,15);
    v16bf a1 = __builtin_shufflevector(l1, h1, 0,1,2,3,4,5,6,7,8,9,10,11,12,13,14,15);
    #pragma unroll
    for (int t = 0; t < 4; ++t) {
      v16bf bfrag = *(const v16bf*)(w + (size_t)(n0 + t * 16 + ln) * D_MODEL + k0 + 16 * g);
      acc[0][t] = __builtin_amdgcn_wmma_f32_16x16x32_bf16(
          false, a0, false, bfrag, (short)0, acc[0][t], false, false);
      acc[1][t] = __builtin_amdgcn_wmma_f32_16x16x32_bf16(
          false, a1, false, bfrag, (short)0, acc[1][t], false, false);
    }
  }

  #pragma unroll
  for (int mi = 0; mi < 2; ++mi) {
    const int mb = m0 + 16 * mi;
    const int b  = mb / SEQ;
    const int s0 = mb % SEQ;
    if (vmode == 0) {
      #pragma unroll
      for (int t = 0; t < 4; ++t) {
        const int c = n0 + t * 16 + ln;
        const int h = c >> 6, d = c & 63;
        __bf16* op = out + ((size_t)(b * NUM_HEADS + h) * SEQ + s0 + 8 * g) * HEAD_DIM + d;
        #pragma unroll
        for (int r = 0; r < 8; ++r) op[(size_t)r * HEAD_DIM] = f2bf(acc[mi][t][r]);
      }
    } else {
      #pragma unroll
      for (int t = 0; t < 4; ++t) {
        const int c = n0 + t * 16 + ln;
        const int h = c >> 6, d = c & 63;
        __bf16* op = out + ((size_t)(b * NUM_HEADS + h) * HEAD_DIM + d) * SEQ + s0 + 8 * g;
        v8bf pk;
        #pragma unroll
        for (int r = 0; r < 8; ++r) pk[r] = f2bf(acc[mi][t][r]);
        *(v8bf*)op = pk;
      }
    }
  }
}

// ---------------------------------------------------------------------------
// Flash attention: block = 8 waves sharing one (b,h) and a 128-row q range.
// K (32x64) and V^T (64x32) chunk tiles are double-buffered through LDS so
// K/V global traffic is shared 8-ways. scores^T = K Q^T, out^T = V^T P.
// ---------------------------------------------------------------------------
__global__ __launch_bounds__(256)
void mha_flash_attn_kernel(const __bf16* __restrict__ Q, const __bf16* __restrict__ K,
                           const __bf16* __restrict__ Vt, __bf16* __restrict__ out) {
  const int tid  = threadIdx.x;
  const int wave = tid >> 5;
  const int lane = tid & 31;
  const int g    = lane >> 4;
  const int ln   = lane & 15;
  const int bh   = blockIdx.x >> 4;          // 32 (b,h) pairs
  const int q0   = (blockIdx.x & 15) * 128 + wave * 16;

  const __bf16* Qp = Q  + (size_t)bh * SEQ * HEAD_DIM;
  const __bf16* Kp = K  + (size_t)bh * SEQ * HEAD_DIM;
  const __bf16* Vp = Vt + (size_t)bh * HEAD_DIM * SEQ;

  __shared__ __bf16 kbuf[2][32 * HEAD_DIM];  // [k-row][d]      4 KB each
  __shared__ __bf16 vbuf[2][HEAD_DIM * 32];  // [d-row][k-col]  4 KB each

  // staging indices (256 threads cover each 4 KB tile with one 16B copy)
  const int krow = tid >> 3, kcol = (tid & 7) * 8;   // 32 x 64
  const int vrow = tid >> 2, vcol = (tid & 3) * 8;   // 64 x 32

  // Q as B operand: halves h -> d = dc + h + 16g, column q = q0 + ln
  v16bf qB[2];
  {
    const __bf16* qrow = Qp + (size_t)(q0 + ln) * HEAD_DIM + 16 * g;
    qB[0] = *(const v16bf*)(qrow);
    qB[1] = *(const v16bf*)(qrow + 32);
  }

  v8f acc[4];
  #pragma unroll
  for (int t = 0; t < 4; ++t)
    #pragma unroll
    for (int r = 0; r < 8; ++r) acc[t][r] = 0.0f;

  float l = 0.0f;
  float mrow = -1e30f;
  const float scale = 0.125f;   // 1/sqrt(64)

  // prologue: stage chunk 0 into buffer 0
  *(v8bf*)(&kbuf[0][krow * HEAD_DIM + kcol]) =
      *(const v8bf*)(Kp + (size_t)krow * HEAD_DIM + kcol);
  *(v8bf*)(&vbuf[0][vrow * 32 + vcol]) =
      *(const v8bf*)(Vp + (size_t)vrow * SEQ + vcol);
  __syncthreads();

  for (int kc = 0; kc < SEQ; kc += 32) {
    const int cur = (kc >> 5) & 1;
    if (kc + 32 < SEQ) {  // stage next chunk into the other buffer
      const int nxt = cur ^ 1;
      const int kn  = kc + 32;
      *(v8bf*)(&kbuf[nxt][krow * HEAD_DIM + kcol]) =
          *(const v8bf*)(Kp + (size_t)(kn + krow) * HEAD_DIM + kcol);
      *(v8bf*)(&vbuf[nxt][vrow * 32 + vcol]) =
          *(const v8bf*)(Vp + (size_t)vrow * SEQ + kn + vcol);
    }

    // scores^T tiles from LDS: T[0] rows k=kc..kc+15, T[1] rows kc+16..kc+31
    v8f T[2];
    #pragma unroll
    for (int t = 0; t < 2; ++t) {
      const __bf16* krp = &kbuf[cur][(16 * t + ln) * HEAD_DIM];
      v8f c = {0.f, 0.f, 0.f, 0.f, 0.f, 0.f, 0.f, 0.f};
      #pragma unroll
      for (int dc = 0; dc < 2; ++dc) {
        v8bf lo = *(const v8bf*)(krp + 32 * dc + 16 * g);
        v8bf hi = *(const v8bf*)(krp + 32 * dc + 16 * g + 8);
        v16bf a = __builtin_shufflevector(lo, hi, 0,1,2,3,4,5,6,7,8,9,10,11,12,13,14,15);
        c = __builtin_amdgcn_wmma_f32_16x16x32_bf16(
            false, a, false, qB[dc], (short)0, c, false, false);
      }
      T[t] = c;
    }

    // online softmax per q column (lane pair lane^16 holds the other 16 k's)
    float mc = -1e30f;
    #pragma unroll
    for (int r = 0; r < 8; ++r) { mc = fmaxf(mc, T[0][r]); mc = fmaxf(mc, T[1][r]); }
    mc *= scale;
    mc = fmaxf(mc, __shfl_xor(mc, 16, 32));
    const float mnew  = fmaxf(mrow, mc);
    const float alpha = __expf(mrow - mnew);

    float p0[8], p1[8], lsum = 0.0f;
    #pragma unroll
    for (int r = 0; r < 8; ++r) {
      p0[r] = __expf(T[0][r] * scale - mnew);
      p1[r] = __expf(T[1][r] * scale - mnew);
      lsum += p0[r] + p1[r];
    }
    lsum += __shfl_xor(lsum, 16, 32);
    l = l * alpha + lsum;
    mrow = mnew;

    #pragma unroll
    for (int t = 0; t < 4; ++t)
      #pragma unroll
      for (int r = 0; r < 8; ++r) acc[t][r] *= alpha;

    // P as B operand (k x q): halves h -> k = kc + h + 16g; one pair exchange
    v16bf pB;
    #pragma unroll
    for (int r = 0; r < 8; ++r) {
      const float e0 = __shfl_xor(p0[r], 16, 32);
      const float e1 = __shfl_xor(p1[r], 16, 32);
      const float lowv = (g == 0) ? p0[r] : e1;
      const float hiv  = (g == 0) ? e0    : p1[r];
      pB[r]     = f2bf(lowv);
      pB[8 + r] = f2bf(hiv);
    }

    // out^T accumulate from LDS: A = V^T tile (16d x 32k), rows d = 16t + ln
    #pragma unroll
    for (int t = 0; t < 4; ++t) {
      const __bf16* vrp = &vbuf[cur][(16 * t + ln) * 32];
      v8bf lo = *(const v8bf*)(vrp + 16 * g);
      v8bf hi = *(const v8bf*)(vrp + 16 * g + 8);
      v16bf a = __builtin_shufflevector(lo, hi, 0,1,2,3,4,5,6,7,8,9,10,11,12,13,14,15);
      acc[t] = __builtin_amdgcn_wmma_f32_16x16x32_bf16(
          false, a, false, pB, (short)0, acc[t], false, false);
    }

    __syncthreads();  // next-chunk staging complete; safe to overwrite cur next iter
  }

  // epilogue: out[b][q0+ln][h*64 + 16t + 8g + r] = acc / l ; packed 16B stores
  const float rl = 1.0f / l;
  const int b = bh >> 4, h = bh & 15;
  __bf16* orow = out + ((size_t)(b * SEQ + q0 + ln) * D_MODEL) + h * HEAD_DIM + 8 * g;
  #pragma unroll
  for (int t = 0; t < 4; ++t) {
    v8bf pk;
    #pragma unroll
    for (int r = 0; r < 8; ++r) pk[r] = f2bf(acc[t][r] * rl);
    *(v8bf*)(orow + 16 * t) = pk;
  }
}

// ---------------------------------------------------------------------------
// Output projection: d_out = attn_out(bf16) @ o_w^T(bf16) + o_b  (f32 result)
// 32x64 strip per wave, B fragments shared across the two m-subtiles.
// ---------------------------------------------------------------------------
__global__ __launch_bounds__(256)
void mha_out_proj_kernel(const __bf16* __restrict__ x, const __bf16* __restrict__ w,
                         const float* __restrict__ bias, float* __restrict__ out) {
  const int lane = threadIdx.x & 31;
  const int g    = lane >> 4;
  const int ln   = lane & 15;
  const int wid  = (blockIdx.x * blockDim.x + threadIdx.x) >> 5;
  const int m0   = (wid >> 4) * 32;
  const int n0   = (wid & 15) * 64;

  v8f acc[2][4];
  #pragma unroll
  for (int t = 0; t < 4; ++t) {
    const float bv = bias[n0 + t * 16 + ln];
    #pragma unroll
    for (int r = 0; r < 8; ++r) { acc[0][t][r] = bv; acc[1][t][r] = bv; }
  }

  const __bf16* xrow0 = x + (size_t)(m0 + ln) * D_MODEL;
  const __bf16* xrow1 = x + (size_t)(m0 + 16 + ln) * D_MODEL;

  for (int k0 = 0; k0 < D_MODEL; k0 += 32) {
    v8bf l0 = *(const v8bf*)(xrow0 + k0 + 16 * g);
    v8bf h0 = *(const v8bf*)(xrow0 + k0 + 16 * g + 8);
    v8bf l1 = *(const v8bf*)(xrow1 + k0 + 16 * g);
    v8bf h1 = *(const v8bf*)(xrow1 + k0 + 16 * g + 8);
    v16bf a0 = __builtin_shufflevector(l0, h0, 0,1,2,3,4,5,6,7,8,9,10,11,12,13,14,15);
    v16bf a1 = __builtin_shufflevector(l1, h1, 0,1,2,3,4,5,6,7,8,9,10,11,12,13,14,15);
    #pragma unroll
    for (int t = 0; t < 4; ++t) {
      v16bf bfrag = *(const v16bf*)(w + (size_t)(n0 + t * 16 + ln) * D_MODEL + k0 + 16 * g);
      acc[0][t] = __builtin_amdgcn_wmma_f32_16x16x32_bf16(
          false, a0, false, bfrag, (short)0, acc[0][t], false, false);
      acc[1][t] = __builtin_amdgcn_wmma_f32_16x16x32_bf16(
          false, a1, false, bfrag, (short)0, acc[1][t], false, false);
    }
  }

  #pragma unroll
  for (int mi = 0; mi < 2; ++mi) {
    #pragma unroll
    for (int t = 0; t < 4; ++t) {
      const int c = n0 + t * 16 + ln;
      float* op = out + (size_t)(m0 + 16 * mi + 8 * g) * D_MODEL + c;
      #pragma unroll
      for (int r = 0; r < 8; ++r) op[(size_t)r * D_MODEL] = acc[mi][t][r];
    }
  }
}

// ---------------------------------------------------------------------------
extern "C" void kernel_launch(void* const* d_in, const int* in_sizes, int n_in,
                              void* d_out, int out_size, void* d_ws, size_t ws_size,
                              hipStream_t stream) {
  (void)in_sizes; (void)n_in; (void)out_size; (void)ws_size;

  const float* query = (const float*)d_in[0];
  const float* key_  = (const float*)d_in[1];
  const float* value = (const float*)d_in[2];
  const float* q_w   = (const float*)d_in[3];
  const float* q_b   = (const float*)d_in[4];
  const float* k_w   = (const float*)d_in[5];
  const float* k_b   = (const float*)d_in[6];
  const float* v_w   = (const float*)d_in[7];
  const float* v_b   = (const float*)d_in[8];
  const float* o_w   = (const float*)d_in[9];
  const float* o_b   = (const float*)d_in[10];
  float* out = (float*)d_out;

  char* ws = (char*)d_ws;
  const size_t SEG  = (size_t)MROWS * D_MODEL * sizeof(__bf16);    // 8 MB
  const size_t WSEG = (size_t)D_MODEL * D_MODEL * sizeof(__bf16);  // 2 MB
  __bf16* qws = (__bf16*)(ws + 0 * SEG);   // Q  [B,H,S,64]
  __bf16* kws = (__bf16*)(ws + 1 * SEG);   // K  [B,H,S,64]
  __bf16* vws = (__bf16*)(ws + 2 * SEG);   // V^T[B,H,64,S]
  __bf16* aws = (__bf16*)(ws + 3 * SEG);   // attn out [B,S,1024]
  __bf16* xq  = (__bf16*)(ws + 4 * SEG);   // bf16 activations
  __bf16* xk  = (__bf16*)(ws + 5 * SEG);
  __bf16* xv  = (__bf16*)(ws + 6 * SEG);
  __bf16* wq  = (__bf16*)(ws + 7 * SEG + 0 * WSEG);
  __bf16* wk  = (__bf16*)(ws + 7 * SEG + 1 * WSEG);
  __bf16* wv  = (__bf16*)(ws + 7 * SEG + 2 * WSEG);
  __bf16* wo  = (__bf16*)(ws + 7 * SEG + 3 * WSEG);

  // conversions: weights 1M elems -> 512 blocks; activations 4M elems -> 2048
  mha_cvt_kernel<<<512,  256, 0, stream>>>(q_w, wq);
  mha_cvt_kernel<<<512,  256, 0, stream>>>(k_w, wk);
  mha_cvt_kernel<<<512,  256, 0, stream>>>(v_w, wv);
  mha_cvt_kernel<<<512,  256, 0, stream>>>(o_w, wo);
  mha_cvt_kernel<<<2048, 256, 0, stream>>>(query, xq);
  mha_cvt_kernel<<<2048, 256, 0, stream>>>(key_,  xk);
  mha_cvt_kernel<<<2048, 256, 0, stream>>>(value, xv);

  // projections: 2048 waves = 128 m-groups(32 rows) x 16 n-strips
  mha_qkv_proj_kernel<<<256, 256, 0, stream>>>(xq, wq, q_b, qws, 0);
  mha_qkv_proj_kernel<<<256, 256, 0, stream>>>(xk, wk, k_b, kws, 0);
  mha_qkv_proj_kernel<<<256, 256, 0, stream>>>(xv, wv, v_b, vws, 1);

  // attention: 512 blocks = 32 (b,h) x 16 q-blocks of 128 rows
  mha_flash_attn_kernel<<<512, 256, 0, stream>>>(qws, kws, vws, aws);

  // output projection
  mha_out_proj_kernel<<<256, 256, 0, stream>>>(aws, wo, o_b, out);
}
